// OriBinLoss_6253472383550
// MI455X (gfx1250) — matrix operand order repaired and checked by reading, not compiled, not run on hardware
//
#include <hip/hip_runtime.h>
#include <hip/hip_bf16.h>
#include <math.h>

// Problem constants (from reference setup)
#define NROW 4096
#define DIM  1024
#define ALPHA  20.0f
#define MARGIN 0.5f

typedef __attribute__((ext_vector_type(16))) _Float16 v16h;
typedef __attribute__((ext_vector_type(8)))  _Float16 v8h;
typedef __attribute__((ext_vector_type(8)))  float    v8f;

// ---------------- helpers ----------------

__device__ __forceinline__ float softplus_f(float x) {
  // log1p(exp(x)), stable
  return (x > 20.0f) ? x : log1pf(expf(x));
}

__device__ __forceinline__ float block_sum(float v, float* sred) {
  const int tid = threadIdx.x;
  sred[tid] = v; __syncthreads();
  #pragma unroll
  for (int s = 128; s > 0; s >>= 1) {
    if (tid < s) sred[tid] += sred[tid + s];
    __syncthreads();
  }
  float r = sred[0]; __syncthreads();
  return r;
}

__device__ __forceinline__ float block_min(float v, float* sred) {
  const int tid = threadIdx.x;
  sred[tid] = v; __syncthreads();
  #pragma unroll
  for (int s = 128; s > 0; s >>= 1) {
    if (tid < s) sred[tid] = fminf(sred[tid], sred[tid + s]);
    __syncthreads();
  }
  float r = sred[0]; __syncthreads();
  return r;
}

__device__ __forceinline__ float block_max(float v, float* sred) {
  const int tid = threadIdx.x;
  sred[tid] = v; __syncthreads();
  #pragma unroll
  for (int s = 128; s > 0; s >>= 1) {
    if (tid < s) sred[tid] = fmaxf(sred[tid], sred[tid + s]);
    __syncthreads();
  }
  float r = sred[0]; __syncthreads();
  return r;
}

// ---------------- pass 0: init global-max accumulator ----------------
__global__ void init_kernel(int* __restrict__ gmax_bits) {
  if (threadIdx.x == 0 && blockIdx.x == 0) {
    gmax_bits[0] = 0;   // float 0.0; true max ~1.0 > 0 so int-compare is valid
  }
}

// ---------------- pass 1: L2 normalize rows, f32 -> f16 ----------------
__global__ void norm_kernel(const float* __restrict__ in, _Float16* __restrict__ x16) {
  __shared__ float sred[256];
  const int row = blockIdx.x;
  const int tid = threadIdx.x;
  const float* r = in + (size_t)row * DIM;
  float ss = 0.0f;
  for (int c = tid; c < DIM; c += 256) { float v = r[c]; ss += v * v; }
  float tot = block_sum(ss, sred);
  float rinv = 1.0f / sqrtf(tot);
  _Float16* o = x16 + (size_t)row * DIM;
  for (int c = tid; c < DIM; c += 256) o[c] = (_Float16)(r[c] * rinv);
}

// ---------------- pass 2: SIM = X * X^T via f16 WMMA, f32 accum ----------------
// Symmetric GEMM: only upper-triangular 64x64 wave tiles are computed; each
// result is stored to (row,col) and mirrored to (col,row). Halves WMMA work
// and L2 fragment traffic.
//
// wave32: each wave computes a 64x64 tile of SIM as 4x4 blocks of 16x16 WMMA.
// A fragment (16x32 f16, contiguous k order, same as B): lane m = lane&15, hi = lane>>4;
//   elems 0..7  = X[m0+m][k0 + hi*16 + 0..7]
//   elems 8..15 = X[m0+m][k0 + hi*16 + 8 + 0..7]
// B fragment (32x16 f16): B[k][n] = X[n0+n][k]; lane n = lane&15, kh = lane>>4;
//   elems 0..15 = X[n0+n][k0 + kh*16 + 0..15]   (contiguous 32B load)
__global__ void __launch_bounds__(256)
gemm_kernel(const _Float16* __restrict__ X, float* __restrict__ SIM,
            int* __restrict__ gmax_bits) {
  const int lane = threadIdx.x & 31;
  const int wave = threadIdx.x >> 5;
  const int t    = blockIdx.x * 8 + wave;     // triangular tile id
  const int TP   = NROW / 64;                 // 64 tiles per dim

  // invert t -> (tm, tn), tm <= tn, row-major over the upper triangle.
  // off(r) = r*(2*TP - r + 1)/2 ; solve off(r) = t.
  int tm = (int)((2.0f * TP + 1.0f -
                  sqrtf((2.0f * TP + 1.0f) * (2.0f * TP + 1.0f) - 8.0f * (float)t)) * 0.5f);
  if (tm < 0) tm = 0;
  while ((tm + 1) * (2 * TP - tm) / 2 <= t) ++tm;         // off(tm+1) <= t -> advance
  while (tm * (2 * TP - tm + 1) / 2 > t) --tm;            // off(tm)   >  t -> retreat
  const int tn = tm + (t - tm * (2 * TP - tm + 1) / 2);

  const int m0 = tm * 64, n0 = tn * 64;

  const int am  = lane & 15, ahi = lane >> 4;
  const int bn  = lane & 15, bkh = lane >> 4;

  v8f acc[4][4] = {};

  const _Float16* arow[4];
  const _Float16* brow[4];
  #pragma unroll
  for (int i = 0; i < 4; ++i) {
    arow[i] = X + (size_t)(m0 + i * 16 + am) * DIM + ahi * 16;
    brow[i] = X + (size_t)(n0 + i * 16 + bn) * DIM + bkh * 16;
  }

  #pragma unroll 2
  for (int k0 = 0; k0 < DIM; k0 += 32) {
    // prefetch next K chunk (high locality -> near scope; dropped past end)
    __builtin_prefetch(arow[0] + k0 + 64, 0, 3);
    __builtin_prefetch(brow[0] + k0 + 64, 0, 3);

    v16h a[4], b[4];
    #pragma unroll
    for (int i = 0; i < 4; ++i) {
      v8h lo = *(const v8h*)(arow[i] + k0);
      v8h hi = *(const v8h*)(arow[i] + k0 + 8);
      #pragma unroll
      for (int e = 0; e < 8; ++e) { a[i][e] = lo[e]; a[i][e + 8] = hi[e]; }
      b[i] = *(const v16h*)(brow[i] + k0);
    }

    #pragma unroll
    for (int i = 0; i < 4; ++i)
      #pragma unroll
      for (int j = 0; j < 4; ++j)
        acc[i][j] = __builtin_amdgcn_wmma_f32_16x16x32_f16(
            false, a[i], false, b[j], (short)0, acc[i][j], false, false);
  }

  // Store C/D per ISA 16x16 f32 layout: lane<16 -> M = r, lane>=16 -> M = r+8;
  // N = lane&15. Mirror-store to fill the lower triangle.
  float vmax = -1e30f;
  const int rOff = (lane >> 4) * 8;
  const int cOff = lane & 15;
  #pragma unroll
  for (int i = 0; i < 4; ++i) {
    #pragma unroll
    for (int j = 0; j < 4; ++j) {
      const int rowBase = m0 + i * 16 + rOff;
      const int colBase = n0 + j * 16 + cOff;
      #pragma unroll
      for (int r = 0; r < 8; ++r) {
        float v = acc[i][j][r];
        vmax = fmaxf(vmax, v);
        SIM[(size_t)(rowBase + r) * NROW + colBase] = v;   // upper
        SIM[(size_t)colBase * NROW + (rowBase + r)] = v;   // mirrored lower
      }
    }
  }
  // wave butterfly max, then one atomic per wave (int compare valid: max > 0)
  #pragma unroll
  for (int off = 16; off > 0; off >>= 1)
    vmax = fmaxf(vmax, __shfl_xor(vmax, off, 32));
  if (lane == 0) atomicMax(gmax_bits, __float_as_int(vmax));
}

// ---------------- pass 3: per-row loss ----------------
__global__ void loss_kernel(const float* __restrict__ SIM, const int* __restrict__ targets,
                            const int* __restrict__ gmax_bits, float* __restrict__ rowloss) {
  __shared__ float sred[256];
  const int i   = blockIdx.x;
  const int tid = threadIdx.x;
  const float gmax = __int_as_float(*gmax_bits);
  const float base = fmaxf(gmax - 0.1f, MARGIN + 0.2f);
  const float* row = SIM + (size_t)i * NROW;
  const int ti = targets[i];

  // Phase A: positives -> min_pos, masked softplus sum (keep: s < base)
  float minp = 1e30f, psum = 0.0f, pcnt = 0.0f;
  for (int j = tid; j < NROW; j += 256) {
    if (j == i) continue;
    if (targets[j] == ti) {
      float s = row[j];
      minp = fminf(minp, s);
      if (s < base) { psum += softplus_f(-2.0f * (s - MARGIN)); pcnt += 1.0f; }
    }
  }
  minp = block_min(minp, sred);
  psum = block_sum(psum, sred);
  pcnt = block_sum(pcnt, sred);

  // Phase B: negatives -> max_neg, masked softplus sum (keep: s > min_pos - 0.05)
  const float thr = minp - 0.05f;
  float maxn = -1e30f, nsum = 0.0f, ncnt = 0.0f;
  for (int j = tid; j < NROW; j += 256) {
    if (targets[j] != ti) {
      float s = row[j];
      maxn = fmaxf(maxn, s);
      if (s > thr) { nsum += softplus_f(ALPHA * (s - MARGIN)); ncnt += 1.0f; }
    }
  }
  maxn = block_max(maxn, sred);
  nsum = block_sum(nsum, sred);
  ncnt = block_sum(ncnt, sred);

  if (tid == 0) {
    float pl = (pcnt > 0.0f) ? (psum / pcnt) : softplus_f(-2.0f * (minp - MARGIN));
    float nl = (ncnt > 0.0f) ? (nsum / ncnt) : softplus_f(ALPHA * (maxn - MARGIN));
    rowloss[i] = pl + nl;
  }
}

// ---------------- pass 4: deterministic mean over rows ----------------
__global__ void finalize_kernel(const float* __restrict__ rowloss, float* __restrict__ out) {
  __shared__ float sred[256];
  const int tid = threadIdx.x;
  float s = 0.0f;
  for (int i = tid; i < NROW; i += 256) s += rowloss[i];   // fixed order per thread
  float tot = block_sum(s, sred);                          // fixed tree order
  if (tid == 0) out[0] = tot * (1.0f / (float)NROW);
}

// ---------------- launch ----------------
extern "C" void kernel_launch(void* const* d_in, const int* in_sizes, int n_in,
                              void* d_out, int out_size, void* d_ws, size_t ws_size,
                              hipStream_t stream) {
  const float* inputs  = (const float*)d_in[0];
  const int*   targets = (const int*)d_in[1];
  float* out = (float*)d_out;

  // workspace layout: [X16: 8MB][SIM: 64MB][rowloss: 16KB][gmax: 4B]
  char* ws = (char*)d_ws;
  _Float16* X16     = (_Float16*)ws;
  float*    SIM     = (float*)(ws + (size_t)NROW * DIM * sizeof(_Float16));
  float*    rowloss = (float*)(ws + (size_t)NROW * DIM * sizeof(_Float16)
                                  + (size_t)NROW * NROW * sizeof(float));
  int* gmax_bits    = (int*)(rowloss + NROW);

  norm_kernel<<<NROW, 256, 0, stream>>>(inputs, X16);
  init_kernel<<<1, 64, 0, stream>>>(gmax_bits);
  // upper-triangular 64x64 tiles: TP*(TP+1)/2 = 64*65/2 = 2080 wave-tiles,
  // 8 waves per 256-thread block -> 260 blocks
  gemm_kernel<<<(NROW / 64) * (NROW / 64 + 1) / 2 / 8, 256, 0, stream>>>(X16, SIM, gmax_bits);
  loss_kernel<<<NROW, 256, 0, stream>>>(SIM, targets, gmax_bits, rowloss);
  finalize_kernel<<<1, 256, 0, stream>>>(rowloss, out);
}
